// IA3Attention_41171556499741
// MI455X (gfx1250) — matrix operand order repaired and checked by reading, not compiled, not run on hardware
//
#include <hip/hip_runtime.h>

#define S_DIM 2048
#define D_DIM 2048
#define H_N   16
#define HD_N  128

typedef __bf16 bf16;
typedef __attribute__((ext_vector_type(16))) __bf16 v16bf;
typedef __attribute__((ext_vector_type(8)))  __bf16 v8bf;
typedef __attribute__((ext_vector_type(8)))  float  v8f;

__device__ __forceinline__ v8f wmma_bf16(v16bf a, v16bf b, v8f c) {
  // D(16x16,f32) = A(16x32,bf16) x B(32x16,bf16) + C
  return __builtin_amdgcn_wmma_f32_16x16x32_bf16(false, a, false, b, (short)0, c,
                                                 false, false);
}

__device__ __forceinline__ v8f vzero8() {
  v8f z;
#pragma unroll
  for (int e = 0; e < 8; ++e) z[e] = 0.0f;
  return z;
}

// ---------------------------------------------------------------- f32 -> bf16
__global__ __launch_bounds__(256) void f2bf_kernel(const float* __restrict__ x,
                                                   bf16* __restrict__ y, int n) {
  int i = blockIdx.x * blockDim.x + threadIdx.x;
  if (i < n) y[i] = (bf16)x[i];
}

// ------------------------------------------------- C[m,n] = sum_k A[m,k]*B[n,k]
// A, B: bf16 [2048 x 2048] row-major; C: f32.
// 8 waves/block as 2(M) x 4(N); wave tile 64x64 = 4x4 WMMA tiles; K step 32.
// Software-pipelined: next K-slab fragments are loaded while the current
// slab's 16 WMMAs run; last iteration peeled (no OOB reads).
__global__ __launch_bounds__(256) void gemm3_nt_kernel(
    const bf16* __restrict__ A,
    const bf16* __restrict__ B0, const bf16* __restrict__ B1,
    const bf16* __restrict__ B2,
    float* __restrict__ C0, float* __restrict__ C1, float* __restrict__ C2) {
  const int lane = threadIdx.x & 31;
  const int wv   = threadIdx.x >> 5;
  const int l15  = lane & 15;
  const int hi   = lane >> 4;   // 0: lanes 0-15, 1: lanes 16-31

  const bf16* Bw = (blockIdx.z == 0) ? B0 : ((blockIdx.z == 1) ? B1 : B2);
  float*      C  = (blockIdx.z == 0) ? C0 : ((blockIdx.z == 1) ? C1 : C2);

  const int m0 = blockIdx.y * 128 + (wv >> 2) * 64;
  const int n0 = blockIdx.x * 256 + (wv & 3) * 64;

  // Fragment loaders.
  // B layout (32x16 bf16): lane holds 16 contiguous k at row (n0+16t+l15),
  //   k offset hi*16.
  // A layout (16x32 bf16): two contiguous 8-elem chunks per lane at
  //   row (m0+16i+l15), k offsets hi*16 and hi*16+8.
  auto loadB = [&](int kb, v16bf bf[4], bool pf) {
#pragma unroll
    for (int t = 0; t < 4; ++t) {
      const bf16* bp = Bw + (size_t)(n0 + t * 16 + l15) * D_DIM + kb + hi * 16;
      if (pf) __builtin_prefetch(bp + 96, 0, 3);  // near-scope, 3 slabs ahead
      bf[t] = *(const v16bf*)bp;
    }
  };
  auto loadA = [&](int kb, v16bf af[4], bool pf) {
#pragma unroll
    for (int i = 0; i < 4; ++i) {
      const bf16* ap = A + (size_t)(m0 + i * 16 + l15) * D_DIM + kb + hi * 16;
      if (pf) __builtin_prefetch(ap + 96, 0, 3);
      union { v16bf v; v8bf h[2]; } u;
      u.h[0] = *(const v8bf*)(ap);
      u.h[1] = *(const v8bf*)(ap + 8);
      af[i] = u.v;
    }
  };

  v8f acc[4][4];
#pragma unroll
  for (int i = 0; i < 4; ++i)
#pragma unroll
    for (int t = 0; t < 4; ++t) acc[i][t] = vzero8();

  v16bf afr[4], bfr[4];
  loadB(0, bfr, true);
  loadA(0, afr, true);

  for (int kb = 0; kb < D_DIM - 32; kb += 32) {
    v16bf an[4], bn[4];
    loadB(kb + 32, bn, true);    // issue next-slab loads first ...
    loadA(kb + 32, an, true);
#pragma unroll
    for (int i = 0; i < 4; ++i)  // ... then 16 independent WMMAs hide them
#pragma unroll
      for (int t = 0; t < 4; ++t) acc[i][t] = wmma_bf16(afr[i], bfr[t], acc[i][t]);
#pragma unroll
    for (int t = 0; t < 4; ++t) { afr[t] = an[t]; bfr[t] = bn[t]; }
  }
#pragma unroll
  for (int i = 0; i < 4; ++i)    // peeled last slab
#pragma unroll
    for (int t = 0; t < 4; ++t) acc[i][t] = wmma_bf16(afr[i], bfr[t], acc[i][t]);

  // C/D layout: VGPR r -> (M = hi*8 + r, N = l15)
#pragma unroll
  for (int i = 0; i < 4; ++i)
#pragma unroll
    for (int t = 0; t < 4; ++t)
#pragma unroll
      for (int r = 0; r < 8; ++r)
        C[(size_t)(m0 + i * 16 + hi * 8 + r) * D_DIM + n0 + t * 16 + l15] =
            acc[i][t][r];
}

// ------------------------------------------- RoPE + IA3 + layout for attention
// Qb,Kb: [H][S][HD] bf16 ; Vt: [H][HD][S] bf16 (transposed for P*V B-frags).
// 1/sqrt(HD) folded into Q.
__global__ __launch_bounds__(256) void rope_ia3_kernel(
    const float* __restrict__ Qr, const float* __restrict__ Kr,
    const float* __restrict__ Vr,
    const float* __restrict__ cosT, const float* __restrict__ sinT,
    const float* __restrict__ lk, const float* __restrict__ lv,
    bf16* __restrict__ Qb, bf16* __restrict__ Kb, bf16* __restrict__ Vt) {
  int idx = blockIdx.x * blockDim.x + threadIdx.x;
  if (idx >= S_DIM * D_DIM) return;
  const int s = idx / D_DIM;
  const int d = idx - s * D_DIM;
  const int h = d / HD_N;
  const int j = d - h * HD_N;
  const int jp  = (j < HD_N / 2) ? (j + HD_N / 2) : (j - HD_N / 2);
  const float sg = (j < HD_N / 2) ? -1.0f : 1.0f;
  const float cv = cosT[s * HD_N + j];
  const float sv = sinT[s * HD_N + j];
  const int base = s * D_DIM + h * HD_N;
  const float q  = Qr[idx], qp = Qr[base + jp];
  const float k  = Kr[idx], kp = Kr[base + jp];
  const float qo = (q * cv + sg * qp * sv) * lk[d] * 0.08838834764831845f;
  const float ko = (k * cv + sg * kp * sv);
  const float vo = Vr[idx] * lv[d];
  Qb[((size_t)h * S_DIM + s) * HD_N + j] = (bf16)qo;
  Kb[((size_t)h * S_DIM + s) * HD_N + j] = (bf16)ko;
  Vt[((size_t)h * HD_N + j) * S_DIM + s] = (bf16)vo;
}

// ------------------------------------------------ causal flash attention, WMMA
// One wave per (head, 16-query-row tile). 32-key steps; per step:
// 8 WMMA for Q*K^T, online softmax, 8 WMMA for P*V. Output bf16 [S][D].
__global__ __launch_bounds__(128) void attn_kernel(
    const bf16* __restrict__ Qb, const bf16* __restrict__ Kb,
    const bf16* __restrict__ Vt, bf16* __restrict__ O) {
  __shared__ bf16 plds[4][16][40];   // per-wave P tile, padded rows
  const int lane  = threadIdx.x & 31;
  const int wv    = threadIdx.x >> 5;
  const int l15   = lane & 15;
  const int hi    = lane >> 4;
  const int h     = blockIdx.y;
  const int qbase = (blockIdx.x * 4 + wv) * 16;

  const bf16* Qh = Qb + (size_t)h * S_DIM * HD_N;
  const bf16* Kh = Kb + (size_t)h * S_DIM * HD_N;
  const bf16* Vh = Vt + (size_t)h * HD_N * S_DIM;

  // Q fragments for all 4 k-chunks of head_dim, kept resident.
  v16bf qf[4];
#pragma unroll
  for (int kc = 0; kc < 4; ++kc) {
    const bf16* qp = Qh + (size_t)(qbase + l15) * HD_N + kc * 32 + hi * 16;
    union { v16bf v; v8bf hh[2]; } u;
    u.hh[0] = *(const v8bf*)(qp);
    u.hh[1] = *(const v8bf*)(qp + 8);
    qf[kc] = u.v;
  }

  float m[8], l[8];
  v8f o[8];
#pragma unroll
  for (int r = 0; r < 8; ++r) { m[r] = -INFINITY; l[r] = 0.0f; }
#pragma unroll
  for (int j = 0; j < 8; ++j) o[j] = vzero8();

  const int kend = qbase + 16;            // causal: keys < qbase+16
  for (int kb = 0; kb < kend; kb += 32) {
    v8f cs[2] = {vzero8(), vzero8()};
#pragma unroll
    for (int kc = 0; kc < 4; ++kc)
#pragma unroll
      for (int t = 0; t < 2; ++t) {
        const bf16* kp = Kh + (size_t)(kb + t * 16 + l15) * HD_N + kc * 32 + hi * 16;
        v16bf bk = *(const v16bf*)kp;
        cs[t] = wmma_bf16(qf[kc], bk, cs[t]);
      }
    if (kb + 31 >= qbase) {               // diagonal step: apply causal mask
#pragma unroll
      for (int t = 0; t < 2; ++t) {
        const int key = kb + t * 16 + l15;
#pragma unroll
        for (int r = 0; r < 8; ++r)
          if (key > qbase + hi * 8 + r) cs[t][r] = -1e30f;
      }
    }
    // row max across 32 columns (lanes within each 16-lane half hold one row)
    float mx[8], srow[8];
#pragma unroll
    for (int r = 0; r < 8; ++r) mx[r] = fmaxf(cs[0][r], cs[1][r]);
#pragma unroll
    for (int off = 1; off <= 8; off <<= 1)
#pragma unroll
      for (int r = 0; r < 8; ++r) mx[r] = fmaxf(mx[r], __shfl_xor(mx[r], off, 32));
#pragma unroll
    for (int r = 0; r < 8; ++r) {
      float nm = fmaxf(m[r], mx[r]);
      float sc = __expf(m[r] - nm);
      m[r] = nm;
      float p0 = __expf(cs[0][r] - nm);
      float p1 = __expf(cs[1][r] - nm);
      cs[0][r] = p0; cs[1][r] = p1;
      srow[r] = p0 + p1;
      l[r] *= sc;
#pragma unroll
      for (int j = 0; j < 8; ++j) o[j][r] *= sc;
    }
#pragma unroll
    for (int off = 1; off <= 8; off <<= 1)
#pragma unroll
      for (int r = 0; r < 8; ++r) srow[r] += __shfl_xor(srow[r], off, 32);
#pragma unroll
    for (int r = 0; r < 8; ++r) l[r] += srow[r];

    // re-lay P from C layout to A-fragment layout through LDS
#pragma unroll
    for (int r = 0; r < 8; ++r) {
      plds[wv][hi * 8 + r][l15]      = (bf16)cs[0][r];
      plds[wv][hi * 8 + r][16 + l15] = (bf16)cs[1][r];
    }
    asm volatile("s_wait_dscnt 0" ::: "memory");
    union { v16bf v; v8bf hh[2]; } up;
    up.hh[0] = *(const v8bf*)&plds[wv][l15][hi * 16];
    up.hh[1] = *(const v8bf*)&plds[wv][l15][hi * 16 + 8];
    v16bf pf = up.v;
    // P(16x32) * V(32x128): B-frags contiguous from transposed V
#pragma unroll
    for (int j = 0; j < 8; ++j) {
      const bf16* vp = Vh + (size_t)(j * 16 + l15) * S_DIM + kb + hi * 16;
      v16bf bv = *(const v16bf*)vp;
      o[j] = wmma_bf16(pf, bv, o[j]);
    }
  }
#pragma unroll
  for (int r = 0; r < 8; ++r) l[r] = 1.0f / l[r];
#pragma unroll
  for (int j = 0; j < 8; ++j)
#pragma unroll
    for (int r = 0; r < 8; ++r)
      O[(size_t)(qbase + hi * 8 + r) * D_DIM + h * HD_N + j * 16 + l15] =
          (bf16)(o[j][r] * l[r]);
}

// ----------------------------------------------------------------------- host
extern "C" void kernel_launch(void* const* d_in, const int* in_sizes, int n_in,
                              void* d_out, int out_size, void* d_ws, size_t ws_size,
                              hipStream_t stream) {
  (void)in_sizes; (void)n_in; (void)out_size; (void)ws_size;
  const float* hidden = (const float*)d_in[0];
  // d_in[1] = attention_mask: exactly the causal mask; implemented analytically.
  const float* cosT = (const float*)d_in[2];
  const float* sinT = (const float*)d_in[3];
  const float* Wq   = (const float*)d_in[4];
  const float* Wk   = (const float*)d_in[5];
  const float* Wv   = (const float*)d_in[6];
  const float* Wo   = (const float*)d_in[7];
  const float* lk   = (const float*)d_in[8];
  const float* lv   = (const float*)d_in[9];
  float* out = (float*)d_out;

  const size_t SD = (size_t)S_DIM * D_DIM;   // 4,194,304
  char*  ws  = (char*)d_ws;
  bf16*  hb  = (bf16*)ws;                    // hidden bf16
  bf16*  Wqb = hb + SD;                      // weights bf16
  bf16*  Wkb = Wqb + SD;
  bf16*  Wvb = Wkb + SD;
  bf16*  Wob = Wvb + SD;
  float* Qr  = (float*)(Wob + SD);           // raw Q/K/V f32
  float* Kr  = Qr + SD;
  float* Vr  = Kr + SD;
  bf16*  Qb  = (bf16*)(Vr + SD);             // Q bf16 [H][S][HD]
  bf16*  Kb  = Qb + SD;                      // K bf16 [H][S][HD]
  bf16*  Vt  = Kb + SD;                      // V bf16 [H][HD][S]
  bf16*  Ab  = Vt + SD;                      // attn out bf16 [S][D]
  // total workspace: 30 * SD bytes ~= 126 MB

  const unsigned nb = (unsigned)(SD / 256);
  f2bf_kernel<<<dim3(nb), dim3(256), 0, stream>>>(hidden, hb, (int)SD);
  f2bf_kernel<<<dim3(nb), dim3(256), 0, stream>>>(Wq, Wqb, (int)SD);
  f2bf_kernel<<<dim3(nb), dim3(256), 0, stream>>>(Wk, Wkb, (int)SD);
  f2bf_kernel<<<dim3(nb), dim3(256), 0, stream>>>(Wv, Wvb, (int)SD);
  f2bf_kernel<<<dim3(nb), dim3(256), 0, stream>>>(Wo, Wob, (int)SD);

  gemm3_nt_kernel<<<dim3(D_DIM / 256, S_DIM / 128, 3), dim3(256), 0, stream>>>(
      hb, Wqb, Wkb, Wvb, Qr, Kr, Vr);
  rope_ia3_kernel<<<dim3(nb), dim3(256), 0, stream>>>(
      Qr, Kr, Vr, cosT, sinT, lk, lv, Qb, Kb, Vt);
  attn_kernel<<<dim3(S_DIM / 64, H_N), dim3(128), 0, stream>>>(Qb, Kb, Vt, Ab);
  gemm3_nt_kernel<<<dim3(D_DIM / 256, S_DIM / 128, 1), dim3(256), 0, stream>>>(
      Ab, Wob, Wob, Wob, out, out, out);
}
